// RBlockPEEfficient_91225105367733
// MI455X (gfx1250) — matrix operand order repaired and checked by reading, not compiled, not run on hardware
//
#include <hip/hip_runtime.h>

// ---------------------------------------------------------------------------
// Shapes (compile-time constants from the reference)
// ---------------------------------------------------------------------------
#define BB   4
#define CC   64
#define FF   64
#define TT   128
#define ICC  32
#define NN   (FF * TT)          // 8192
#define MM   ((FF/2) * (TT/2))  // 2048
#define MCHUNK 256
#define NCHUNKS (MM / MCHUNK)   // 8

typedef __attribute__((ext_vector_type(16))) __bf16          v16bf;
typedef __attribute__((ext_vector_type(8)))  unsigned int    v8u;
typedef __attribute__((ext_vector_type(8)))  float           v8f;

__device__ __forceinline__ unsigned short f2bf(float f) {
    unsigned int b = __float_as_uint(f);
    unsigned int r = (b + 0x7FFFu + ((b >> 16) & 1u)) >> 16;   // RNE
    return (unsigned short)r;
}

// Packed bf16 relu: one V_PK_MAX_NUM_BF16 per bf16 pair (ISA VOP3P op 44).
__device__ __forceinline__ unsigned int pk_relu_bf16(unsigned int s) {
    unsigned int d;
    asm("v_pk_max_num_bf16 %0, %1, 0" : "=v"(d) : "v"(s));
    return d;
}

// Async global -> LDS copies (CDNA5 ASYNCcnt path), with safe fallbacks.
// Builtin signatures (from hipcc diagnostics): arg0 = typed pointer in AS(1)
// (global), arg1 = typed pointer in AS(3) (LDS), then imm offset, imm cpol.
typedef int v4i_vs __attribute__((vector_size(16)));
typedef __attribute__((address_space(1))) v4i_vs* g128p;
typedef __attribute__((address_space(3))) v4i_vs* l128p;

__device__ __forceinline__ void async_ld_b128(const void* g, void* l) {
#if __has_builtin(__builtin_amdgcn_global_load_async_to_lds_b128)
    __builtin_amdgcn_global_load_async_to_lds_b128((g128p)(void*)g, (l128p)l, 0, 0);
#else
    *(uint4*)l = *(const uint4*)g;
#endif
}

__device__ __forceinline__ void wait_async() {
#if __has_builtin(__builtin_amdgcn_global_load_async_to_lds_b128)
#if __has_builtin(__builtin_amdgcn_s_wait_asynccnt)
    __builtin_amdgcn_s_wait_asynccnt(0);
#else
    asm volatile("s_wait_asynccnt 0x0" ::: "memory");
#endif
#endif
}

// ---------------------------------------------------------------------------
// Prep kernel 1: u[b,n] = (h0^T theta_w) . x[b,:,n] + h0.theta_b + pos_a(n)
// ---------------------------------------------------------------------------
__global__ void __launch_bounds__(256)
prep_u_kernel(const float* __restrict__ x, const float* __restrict__ theta_w,
              const float* __restrict__ theta_b, const float* __restrict__ h0_w,
              const float* __restrict__ h2_w, float* __restrict__ u) {
    __shared__ float te_w[CC];
    __shared__ float te_b;
    int t = threadIdx.x;
    if (t < CC) {
        float s = 0.f;
        for (int c = 0; c < ICC; ++c) s += h0_w[c] * theta_w[c * CC + t];
        te_w[t] = s;
    }
    if (t == CC) {
        float s = 0.f;
        for (int c = 0; c < ICC; ++c) s += h0_w[c] * theta_b[c];
        te_b = s;
    }
    __syncthreads();
    int gid = blockIdx.x * 256 + t;        // 0 .. B*N-1
    int b = gid / NN, n = gid % NN;
    const float* xb = x + (size_t)b * CC * NN + n;
    float acc = te_b;
    #pragma unroll 8
    for (int ch = 0; ch < CC; ++ch) acc += te_w[ch] * xb[(size_t)ch * NN];
    int fr = n / TT, tc = n % TT;
    float a = h2_w[0] * (fr * (1.0f / (FF - 1))) + h2_w[1] * (tc * (1.0f / (TT - 1)));
    u[gid] = acc + a;
}

// ---------------------------------------------------------------------------
// Prep kernel 2: per (b,m):
//   psi_t[b][c][m] = bf16( maxpool2(psi conv)[c,m] / M )     (transposed layout)
//   v_bf[b,m]      = bf16( sum_c h1[c]*maxpool2(phi conv)[c,m] - pos_b(m) )
// ---------------------------------------------------------------------------
__global__ void __launch_bounds__(256)
prep_vpsi_kernel(const float* __restrict__ x,
                 const float* __restrict__ psi_w, const float* __restrict__ psi_b,
                 const float* __restrict__ phi_w, const float* __restrict__ phi_b,
                 const float* __restrict__ h1_w,  const float* __restrict__ h2_w,
                 unsigned short* __restrict__ v_bf, unsigned short* __restrict__ psi_t) {
    int gid = blockIdx.x * 256 + threadIdx.x;   // 0 .. B*M-1
    int b = gid / MM, m = gid % MM;
    int fm = m / (TT / 2), tm = m % (TT / 2);
    const float* xb = x + (size_t)b * CC * NN;
    int base00 = (2 * fm) * TT + 2 * tm;
    float vacc = 0.f;
    for (int c = 0; c < ICC; ++c) {
        float pb = psi_b[c], qb = phi_b[c];
        float p00 = pb, p01 = pb, p10 = pb, p11 = pb;
        float q00 = qb, q01 = qb, q10 = qb, q11 = qb;
        const float* pw = psi_w + c * CC;
        const float* qw = phi_w + c * CC;
        for (int ch = 0; ch < CC; ++ch) {
            const float* xp = xb + (size_t)ch * NN + base00;
            float x00 = xp[0], x01 = xp[1], x10 = xp[TT], x11 = xp[TT + 1];
            float wp = pw[ch], wq = qw[ch];
            p00 += wp * x00; p01 += wp * x01; p10 += wp * x10; p11 += wp * x11;
            q00 += wq * x00; q01 += wq * x01; q10 += wq * x10; q11 += wq * x11;
        }
        float pmax = fmaxf(fmaxf(p00, p01), fmaxf(p10, p11));
        float qmax = fmaxf(fmaxf(q00, q01), fmaxf(q10, q11));
        psi_t[((size_t)b * ICC + c) * MM + m] = f2bf(pmax * (1.0f / MM));
        vacc += h1_w[c] * qmax;
    }
    float bp = h2_w[0] * (fm * (1.0f / (FF / 2 - 1))) + h2_w[1] * (tm * (1.0f / (TT / 2 - 1)));
    v_bf[gid] = f2bf(vacc - bp);
}

// ---------------------------------------------------------------------------
// Main kernel: fused attention GEMM (WMMA bf16) + W conv + BN + residual.
// Grid: (N/128, B) blocks of 256 threads (8 waves). Wave w owns n-rows
// [nb+16w, nb+16w+16) x all 32 IC columns (two 16x16 WMMA accumulators).
// LDS double-buffered; psi/v chunks prefetched with async-to-LDS loads.
// v is staged pre-swizzled into contiguous-K A order: 8-blocks 0,1 of each 32 K -> vA
// (lane-half 0), 8-blocks 2,3 -> vB (lane-half 1), so each lane's 16 A-side
// v values are one contiguous 32 B LDS read. relu(u+v) runs fully packed:
// v_pk_add_bf16 + v_pk_max_num_bf16 -- no f32<->bf16 cvt in the hot loop.
// ---------------------------------------------------------------------------
__global__ void __launch_bounds__(256)
attn_main_kernel(const float* __restrict__ x, const float* __restrict__ u,
                 const unsigned short* __restrict__ v_bf,
                 const unsigned short* __restrict__ psi_t,
                 const float* __restrict__ W_w, const float* __restrict__ W_b,
                 const float* __restrict__ bn_gamma, const float* __restrict__ bn_beta,
                 const float* __restrict__ bn_mean,  const float* __restrict__ bn_var,
                 float* __restrict__ out) {
    __shared__ alignas(32) unsigned short psi_s[2][ICC * MCHUNK];  // 2 x 16 KB bf16
    __shared__ alignas(32) unsigned short vA_s[2][MCHUNK / 2];     // 8-blocks 0,1 of each 4
    __shared__ alignas(32) unsigned short vB_s[2][MCHUNK / 2];     // 8-blocks 2,3 of each 4
    __shared__ float y_s[8][16][ICC + 1];                          // ~17 KB

    const int b  = blockIdx.y;
    const int nb = blockIdx.x * 128;
    const int t  = threadIdx.x;
    const int wv = t >> 5;          // wave id 0..7
    const int l  = t & 31;          // lane
    const int lrow = l & 15;
    const int lhi  = l >> 4;        // 0/1: which half-wave

    const float u_l = u[b * NN + nb + wv * 16 + lrow];
    const __bf16 u_h = __builtin_bit_cast(__bf16, f2bf(u_l));
    const v16bf uu = (v16bf)u_h;    // splat (feeds v_pk_add_bf16)

    const unsigned short* psi_base = psi_t + (size_t)b * ICC * MM;
    const unsigned short* v_base   = v_bf + (size_t)b * MM;

    // Per-thread psi staging: c = t/8, 32 bf16 (64 B) segment per thread.
    const int sc   = t >> 3;
    const int sseg = (t & 7) * 32;
    // v staging: threads 0..31 each move one 8-bf16 block, swizzled A-order.
    const int vblk = t;                      // block id 0..31 (threads < 32)
    const int vhalf = (vblk >> 1) & 1;       // 0 -> vA, 1 -> vB
    const int vdst  = (vblk >> 2) * 16 + (vblk & 1) * 8;

    // Prefetch chunk 0 into buffer 0.
    {
        const unsigned short* src = psi_base + (size_t)sc * MM + 0 + sseg;
        unsigned short* dst = &psi_s[0][sc * MCHUNK + sseg];
        async_ld_b128(src,      dst);
        async_ld_b128(src + 8,  dst + 8);
        async_ld_b128(src + 16, dst + 16);
        async_ld_b128(src + 24, dst + 24);
        if (t < 32)
            async_ld_b128(v_base + vblk * 8,
                          vhalf ? &vB_s[0][vdst] : &vA_s[0][vdst]);
    }

    v8f acc0 = {}; v8f acc1 = {};

    for (int ci = 0; ci < NCHUNKS; ++ci) {
        const int buf = ci & 1;
        wait_async();           // my chunk-ci async copies have landed in LDS
        __syncthreads();        // everyone's have (and prev compute is done)

        if (ci + 1 < NCHUNKS) { // prefetch next chunk into the other buffer
            const int mc2 = (ci + 1) * MCHUNK;
            const unsigned short* src = psi_base + (size_t)sc * MM + mc2 + sseg;
            unsigned short* dst = &psi_s[buf ^ 1][sc * MCHUNK + sseg];
            async_ld_b128(src,      dst);
            async_ld_b128(src + 8,  dst + 8);
            async_ld_b128(src + 16, dst + 16);
            async_ld_b128(src + 24, dst + 24);
            if (t < 32)
                async_ld_b128(v_base + mc2 + vblk * 8,
                              vhalf ? &vB_s[buf ^ 1][vdst] : &vA_s[buf ^ 1][vdst]);
        }

        const unsigned short* vp = lhi ? vB_s[buf] : vA_s[buf];

        #pragma unroll
        for (int sub = 0; sub < MCHUNK / 32; ++sub) {
            // A tile (16x32 bf16): f = relu(u+v), fully packed bf16.
            v16bf vv = *(const v16bf*)&vp[sub * 16];
            v8u  su = __builtin_bit_cast(v8u, vv + uu);   // 8x v_pk_add_bf16
            v8u  aw;
            #pragma unroll
            for (int j = 0; j < 8; ++j) aw[j] = pk_relu_bf16(su[j]);
            v16bf a = __builtin_bit_cast(v16bf, aw);
            // B tiles (32x16 bf16): lane l holds column lrow, K in
            // [lhi*16, lhi*16+16) contiguous -> two ds_load_b128 each
            const int moff = sub * 32 + lhi * 16;
            v16bf b0 = *(const v16bf*)&psi_s[buf][lrow        * MCHUNK + moff];
            v16bf b1 = *(const v16bf*)&psi_s[buf][(16 + lrow) * MCHUNK + moff];
            acc0 = __builtin_amdgcn_wmma_f32_16x16x32_bf16(false, a, false, b0,
                                                           (short)0, acc0, false, false);
            acc1 = __builtin_amdgcn_wmma_f32_16x16x32_bf16(false, a, false, b1,
                                                           (short)0, acc1, false, false);
        }
    }

    // Transpose accumulators into LDS: D element r -> row r+8*lhi, col lrow(+16)
    __syncthreads();
    #pragma unroll
    for (int r = 0; r < 8; ++r) {
        y_s[wv][r + 8 * lhi][lrow]      = acc0[r];
        y_s[wv][r + 8 * lhi][16 + lrow] = acc1[r];
    }
    __syncthreads();

    // Epilogue: out[b,o,n] = scale*( W[o,:32].y[:,n] + W_b[o] ) + shift + x[b,o,n]
    // lane covers n = its row, o in [lhi*32, lhi*32+32)
    float yr[ICC];
    #pragma unroll
    for (int c = 0; c < ICC; ++c) yr[c] = y_s[wv][lrow][c];

    const int nglob = nb + wv * 16 + lrow;
    const float* xcol = x   + (size_t)b * CC * NN + nglob;
    float*       ocol = out + (size_t)b * CC * NN + nglob;

    for (int oo = 0; oo < 32; ++oo) {
        int o = lhi * 32 + oo;
        const float* wrow = W_w + o * (ICC + 1);   // last entry hits zero channel
        float dot = W_b[o];
        #pragma unroll
        for (int c = 0; c < ICC; ++c) dot += wrow[c] * yr[c];
        float inv = bn_gamma[o] * __frsqrt_rn(bn_var[o] + 1e-5f);
        ocol[(size_t)o * NN] = dot * inv + (bn_beta[o] - bn_mean[o] * inv)
                               + xcol[(size_t)o * NN];
    }
}

// ---------------------------------------------------------------------------
extern "C" void kernel_launch(void* const* d_in, const int* in_sizes, int n_in,
                              void* d_out, int out_size, void* d_ws, size_t ws_size,
                              hipStream_t stream) {
    const float* x        = (const float*)d_in[0];
    const float* psi_w    = (const float*)d_in[1];
    const float* psi_b    = (const float*)d_in[2];
    const float* theta_w  = (const float*)d_in[3];
    const float* theta_b  = (const float*)d_in[4];
    const float* phi_w    = (const float*)d_in[5];
    const float* phi_b    = (const float*)d_in[6];
    const float* h0_w     = (const float*)d_in[7];
    const float* h1_w     = (const float*)d_in[8];
    const float* h2_w     = (const float*)d_in[9];
    const float* W_w      = (const float*)d_in[10];
    const float* W_b      = (const float*)d_in[11];
    const float* bn_gamma = (const float*)d_in[12];
    const float* bn_beta  = (const float*)d_in[13];
    const float* bn_mean  = (const float*)d_in[14];
    const float* bn_var   = (const float*)d_in[15];
    float* out = (float*)d_out;

    // Workspace layout: u (B*N f32) | v_bf (B*M u16) | psi_t (B*IC*M u16)
    float* u_ws = (float*)d_ws;
    unsigned short* v_ws   = (unsigned short*)(u_ws + BB * NN);
    unsigned short* psi_ws = v_ws + BB * MM;

    prep_u_kernel<<<(BB * NN) / 256, 256, 0, stream>>>(x, theta_w, theta_b, h0_w, h2_w, u_ws);
    prep_vpsi_kernel<<<(BB * MM) / 256, 256, 0, stream>>>(x, psi_w, psi_b, phi_w, phi_b,
                                                          h1_w, h2_w, v_ws, psi_ws);
    attn_main_kernel<<<dim3(NN / 128, BB), 256, 0, stream>>>(x, u_ws, v_ws, psi_ws,
                                                             W_w, W_b, bn_gamma, bn_beta,
                                                             bn_mean, bn_var, out);
}
